// Con1D_to_Capsule_36103495090670
// MI455X (gfx1250) — compile-verified, matrix operand order checked by reading, not run on hardware
//
#include <hip/hip_runtime.h>
#include <hip/hip_bf16.h>
#include <math.h>

typedef __attribute__((ext_vector_type(16))) __bf16 v16bf;
typedef __attribute__((ext_vector_type(8)))  float  v8f;

#define B_BATCH 64
#define C_IN    256
#define T_LEN   1024
#define N_CAPS  32
#define D_CAPS  16
#define NO      (N_CAPS*D_CAPS)   /* 512 */
#define TCH     8                 /* t-chunks per batch for routing partials */
#define TCHLEN  (T_LEN/TCH)       /* 128 */

/* ---------------- GEMM: u_hat[b] = x[b]^T (T x C)  *  W (C x NO) ---------- */
/* bf16x3 split precision on v_wmma_f32_16x16x32_bf16, fp32 accumulate.       */
/* Register double-buffered global->LDS staging.                              */

#define BM 128
#define BN 128
#define BK 32
#define LDSA 132   /* padded row stride (dwords) to avoid bank conflicts */

__global__ __launch_bounds__(256)
void caps_gemm_bf16x3(const float* __restrict__ x,
                      const float* __restrict__ W,
                      float* __restrict__ u_hat) {
  const int b    = blockIdx.z;
  const int t0   = blockIdx.y * BM;
  const int o0   = blockIdx.x * BN;
  const int tid  = threadIdx.x;
  const int lane = tid & 31;
  const int wave = tid >> 5;
  const int lh   = lane & 15;   /* m / n within 16x16 tile  */
  const int kh   = lane >> 4;   /* K-half selector          */

  /* 8 waves: 2 (M) x 4 (N); each wave owns a 64x32 output tile */
  const int wm0 = (wave >> 2) * 64;
  const int wn0 = (wave & 3) * 32;

  __shared__ float lds_a[BK * LDSA];   /* [k][m] : A^T tile, m-contiguous  */
  __shared__ float lds_b[BK * LDSA];   /* [k][n] : W tile,   n-contiguous  */

  const float* xb = x + (size_t)b * C_IN * T_LEN;

  /* per-thread staging slots (4 float4 rows of the 32x128 tiles) */
  int skk[4], smm[4];
#pragma unroll
  for (int i = 0; i < 4; ++i) {
    const int slot = tid + i * 256;   /* 1024 float4 slots, 32 per k-row */
    skk[i] = slot >> 5;
    smm[i] = (slot & 31) << 2;
  }

  v8f acc[4][2] = {};
  float4 va[4], vb[4];

  /* preload k0 = 0 */
#pragma unroll
  for (int i = 0; i < 4; ++i) {
    va[i] = *(const float4*)(xb + (size_t)skk[i] * T_LEN + t0 + smm[i]);
    vb[i] = *(const float4*)(W + (size_t)skk[i] * NO + o0 + smm[i]);
  }
#pragma unroll
  for (int i = 0; i < 4; ++i) {
    lds_a[skk[i] * LDSA + smm[i] + 0] = va[i].x;
    lds_a[skk[i] * LDSA + smm[i] + 1] = va[i].y;
    lds_a[skk[i] * LDSA + smm[i] + 2] = va[i].z;
    lds_a[skk[i] * LDSA + smm[i] + 3] = va[i].w;
    lds_b[skk[i] * LDSA + smm[i] + 0] = vb[i].x;
    lds_b[skk[i] * LDSA + smm[i] + 1] = vb[i].y;
    lds_b[skk[i] * LDSA + smm[i] + 2] = vb[i].z;
    lds_b[skk[i] * LDSA + smm[i] + 3] = vb[i].w;
  }
  __syncthreads();

  for (int k0 = 0; k0 < C_IN; k0 += BK) {
    const bool has_next = (k0 + BK) < C_IN;
    /* ---- issue next tile's global loads early (double buffer in regs) ---- */
    if (has_next) {
#pragma unroll
      for (int i = 0; i < 4; ++i) {
        va[i] = *(const float4*)(xb + (size_t)(k0 + BK + skk[i]) * T_LEN + t0 + smm[i]);
        vb[i] = *(const float4*)(W + (size_t)(k0 + BK + skk[i]) * NO + o0 + smm[i]);
      }
    }

    /* ---- build B fragments (hi + residual) for both 16-wide N subtiles ---- */
    v16bf bhi[2], blo[2];
#pragma unroll
    for (int ns = 0; ns < 2; ++ns) {
      const int n = wn0 + ns * 16 + lh;
#pragma unroll
      for (int j = 0; j < 8; ++j) {
        const int kk = 2 * (j & 3) + 8 * kh + 16 * (j >> 2);
        const float f0 = lds_b[kk * LDSA + n];
        const float f1 = lds_b[(kk + 1) * LDSA + n];
        const __bf16 h0 = (__bf16)f0;
        const __bf16 h1 = (__bf16)f1;
        bhi[ns][2 * j + 0] = h0;
        bhi[ns][2 * j + 1] = h1;
        blo[ns][2 * j + 0] = (__bf16)(f0 - (float)h0);
        blo[ns][2 * j + 1] = (__bf16)(f1 - (float)h1);
      }
    }

    /* ---- per M-subtile: build A fragments, 3 WMMAs per (ms,ns) ---- */
#pragma unroll
    for (int ms = 0; ms < 4; ++ms) {
      const int m = wm0 + ms * 16 + lh;
      v16bf ahi, alo;
#pragma unroll
      for (int j = 0; j < 8; ++j) {
        const int kk = 2 * (j & 3) + 8 * kh + 16 * (j >> 2);
        const float f0 = lds_a[kk * LDSA + m];
        const float f1 = lds_a[(kk + 1) * LDSA + m];
        const __bf16 h0 = (__bf16)f0;
        const __bf16 h1 = (__bf16)f1;
        ahi[2 * j + 0] = h0;
        ahi[2 * j + 1] = h1;
        alo[2 * j + 0] = (__bf16)(f0 - (float)h0);
        alo[2 * j + 1] = (__bf16)(f1 - (float)h1);
      }
#pragma unroll
      for (int ns = 0; ns < 2; ++ns) {
        acc[ms][ns] = __builtin_amdgcn_wmma_f32_16x16x32_bf16(
            false, ahi, false, bhi[ns], (short)0, acc[ms][ns], false, false);
        acc[ms][ns] = __builtin_amdgcn_wmma_f32_16x16x32_bf16(
            false, alo, false, bhi[ns], (short)0, acc[ms][ns], false, false);
        acc[ms][ns] = __builtin_amdgcn_wmma_f32_16x16x32_bf16(
            false, ahi, false, blo[ns], (short)0, acc[ms][ns], false, false);
      }
    }

    if (has_next) {
      __syncthreads();
#pragma unroll
      for (int i = 0; i < 4; ++i) {
        lds_a[skk[i] * LDSA + smm[i] + 0] = va[i].x;
        lds_a[skk[i] * LDSA + smm[i] + 1] = va[i].y;
        lds_a[skk[i] * LDSA + smm[i] + 2] = va[i].z;
        lds_a[skk[i] * LDSA + smm[i] + 3] = va[i].w;
        lds_b[skk[i] * LDSA + smm[i] + 0] = vb[i].x;
        lds_b[skk[i] * LDSA + smm[i] + 1] = vb[i].y;
        lds_b[skk[i] * LDSA + smm[i] + 2] = vb[i].z;
        lds_b[skk[i] * LDSA + smm[i] + 3] = vb[i].w;
      }
      __syncthreads();
    }
  }

  /* ---- epilogue: C/D layout VGPR r -> M = r + 8*kh, N = lh ---- */
#pragma unroll
  for (int ms = 0; ms < 4; ++ms) {
#pragma unroll
    for (int ns = 0; ns < 2; ++ns) {
#pragma unroll
      for (int r = 0; r < 8; ++r) {
        const int t = t0 + wm0 + ms * 16 + r + 8 * kh;
        const int o = o0 + wn0 + ns * 16 + lh;
        u_hat[((size_t)b * T_LEN + t) * NO + o] = acc[ms][ns][r];
      }
    }
  }
}

/* ------------- Routing phase A: per (batch, t-chunk) partial s[n,d] ------- */
/* b-logits recomputed as dot(Vsum, u_hat) => no (B,N,T) tensor needed.       */
/* lane == capsule n, so the 32-way softmax is a pure wave32 shuffle reduce.  */

__global__ __launch_bounds__(512)
void caps_route_partial(const float* __restrict__ u_hat,
                        const float* __restrict__ Vsum,
                        float* __restrict__ part, int iter) {
  const int tc   = blockIdx.x;       /* 0..TCH-1 */
  const int b    = blockIdx.y;
  const int tid  = threadIdx.x;      /* 0..511   */
  const int wave = tid >> 5;         /* 0..15    */
  const int lane = tid & 31;         /* == capsule index n */

  __shared__ float red[16 * 512];

  float Vs[D_CAPS];
  if (iter == 0) {
#pragma unroll
    for (int d = 0; d < D_CAPS; ++d) Vs[d] = 0.0f;
  } else {
    const float4* vp = (const float4*)(Vsum + (size_t)b * NO + lane * D_CAPS);
    const float4 p0 = vp[0], p1 = vp[1], p2 = vp[2], p3 = vp[3];
    Vs[0] = p0.x; Vs[1] = p0.y; Vs[2] = p0.z; Vs[3] = p0.w;
    Vs[4] = p1.x; Vs[5] = p1.y; Vs[6] = p1.z; Vs[7] = p1.w;
    Vs[8] = p2.x; Vs[9] = p2.y; Vs[10] = p2.z; Vs[11] = p2.w;
    Vs[12] = p3.x; Vs[13] = p3.y; Vs[14] = p3.z; Vs[15] = p3.w;
  }

  float sloc[D_CAPS];
#pragma unroll
  for (int d = 0; d < D_CAPS; ++d) sloc[d] = 0.0f;

  const float* ub = u_hat + ((size_t)b * T_LEN + (size_t)tc * TCHLEN) * NO;

#pragma unroll
  for (int i = 0; i < TCHLEN / 16; ++i) {     /* 8 t per wave */
    const int t = wave + i * 16;
    const float4* up = (const float4*)(ub + (size_t)t * NO + lane * D_CAPS);
    const float4 q0 = up[0], q1 = up[1], q2 = up[2], q3 = up[3];
    const float u[D_CAPS] = {q0.x, q0.y, q0.z, q0.w, q1.x, q1.y, q1.z, q1.w,
                             q2.x, q2.y, q2.z, q2.w, q3.x, q3.y, q3.z, q3.w};
    float bn = 0.0f;
#pragma unroll
    for (int d = 0; d < D_CAPS; ++d) bn = fmaf(Vs[d], u[d], bn);

    /* softmax over the 32 capsules == full-wave reduction */
    float mx = bn;
    mx = fmaxf(mx, __shfl_xor(mx, 16, 32));
    mx = fmaxf(mx, __shfl_xor(mx, 8, 32));
    mx = fmaxf(mx, __shfl_xor(mx, 4, 32));
    mx = fmaxf(mx, __shfl_xor(mx, 2, 32));
    mx = fmaxf(mx, __shfl_xor(mx, 1, 32));
    float e = __expf(bn - mx);
    float sum = e;
    sum += __shfl_xor(sum, 16, 32);
    sum += __shfl_xor(sum, 8, 32);
    sum += __shfl_xor(sum, 4, 32);
    sum += __shfl_xor(sum, 2, 32);
    sum += __shfl_xor(sum, 1, 32);
    const float c = e / sum;
#pragma unroll
    for (int d = 0; d < D_CAPS; ++d) sloc[d] = fmaf(c, u[d], sloc[d]);
  }

  /* cross-wave serial reduction in LDS (conflict-free strided reads) */
#pragma unroll
  for (int d = 0; d < D_CAPS; ++d)
    red[wave * 512 + lane * D_CAPS + d] = sloc[d];
  __syncthreads();

  float s = 0.0f;                   /* thread tid owns flat (n,d) index tid */
#pragma unroll
  for (int w = 0; w < 16; ++w) s += red[w * 512 + tid];

  part[((size_t)b * TCH + tc) * NO + tid] = s;
}

/* ------------- Routing phase B: fold partials, squash, update Vsum -------- */

__global__ __launch_bounds__(512)
void caps_route_reduce(const float* __restrict__ part,
                       float* __restrict__ Vsum,
                       float* __restrict__ out, int iter) {
  const int b   = blockIdx.x;
  const int tid = threadIdx.x;      /* flat (n = tid>>4, d = tid&15) */

  float s = 0.0f;
#pragma unroll
  for (int tc = 0; tc < TCH; ++tc)
    s += part[((size_t)b * TCH + tc) * NO + tid];

  /* squash: ||s||^2 over the 16-lane d-group (groups align with halves) */
  float s2 = s * s;
  s2 += __shfl_xor(s2, 8, 32);
  s2 += __shfl_xor(s2, 4, 32);
  s2 += __shfl_xor(s2, 2, 32);
  s2 += __shfl_xor(s2, 1, 32);
  const float tot   = s2 + 1e-7f;
  const float scale = sqrtf(tot) / (1.0f + tot);
  const float v     = scale * s;

  const size_t idx = (size_t)b * NO + tid;
  if (iter == 0) Vsum[idx] = v;       /* no pre-zeroing needed vs poison */
  else           Vsum[idx] += v;
  if (iter == 2) out[idx] = v;        /* (B,N,D) flat */
}

extern "C" void kernel_launch(void* const* d_in, const int* in_sizes, int n_in,
                              void* d_out, int out_size, void* d_ws, size_t ws_size,
                              hipStream_t stream) {
  const float* x = (const float*)d_in[0];
  const float* W = (const float*)d_in[1];
  float* out     = (float*)d_out;

  float* u_hat = (float*)d_ws;                               /* 128 MB      */
  float* part  = u_hat + (size_t)B_BATCH * T_LEN * NO;       /* +1 MB       */
  float* Vsum  = part + (size_t)B_BATCH * TCH * NO;          /* +128 KB     */

  dim3 ggrid(NO / BN, T_LEN / BM, B_BATCH);                  /* 4 x 8 x 64  */
  caps_gemm_bf16x3<<<ggrid, 256, 0, stream>>>(x, W, u_hat);

  dim3 rgrid(TCH, B_BATCH);                                  /* 512 WGs     */
  for (int it = 0; it < 3; ++it) {
    caps_route_partial<<<rgrid, 512, 0, stream>>>(u_hat, Vsum, part, it);
    caps_route_reduce<<<B_BATCH, 512, 0, stream>>>(part, Vsum, out, it);
  }
}
